// GAT_65463891526243
// MI455X (gfx1250) — matrix operand order repaired and checked by reading, not compiled, not run on hardware
//
#include <hip/hip_runtime.h>
#include <stdint.h>

typedef __attribute__((ext_vector_type(16))) __bf16 v16bf;
typedef __attribute__((ext_vector_type(8)))  float  v8f;
typedef unsigned int u32x4 __attribute__((ext_vector_type(4)));
typedef int          i32x4 __attribute__((ext_vector_type(4)));
typedef int          i32x8 __attribute__((ext_vector_type(8)));

#define ALPHA 0.2f
#define NB   8
#define NN   2048
#define FD   256     // layer in/out feature dim
#define DH   128     // per-head dim
#define ROWS (NB*NN) // 16384

// TDM-staged B tile: 128 rows x 32 cols bf16, pad 16B after each 64B row -> 80B pitch
// (row pitch 80B keeps every row 16B-aligned for ds_load_b128 and breaks the
//  4-way bank conflict a 64B pitch would give across the 16 d-rows).
#define TILE_PITCH_B 80
#define TILE_PITCH_H 40                    // pitch in ushort elements
#define BUF_BYTES (DH * TILE_PITCH_B)      // 10240
#define SR_BYTES  (NN * 4)                 // 8192

#if __has_include(<hip/amd_detail/amd_gfx1250_TDM.h>)
#define TDM_6ARG 1
#else
#define TDM_6ARG 0
#endif

__device__ __forceinline__ unsigned short f2bf(float f) {
  unsigned u = __float_as_uint(f);
  unsigned r = (u + 0x7FFFu + ((u >> 16) & 1u)) >> 16;  // RNE
  return (unsigned short)r;
}
__device__ __forceinline__ float lrelu(float x) { return x > 0.f ? x : ALPHA * x; }

union ABu { v16bf v; uint4 q[2]; unsigned short s[16]; };

// One TDM load: 2D tile (32 cols x 128 rows) of bf16 from global (row stride
// `stride` elements, dim0 length `dim0` elements) into LDS at lds_addr,
// padding 4 dwords per 16 dwords stored.
__device__ __forceinline__ void tdm_tile_load(const unsigned short* gptr,
                                              unsigned dim0, unsigned stride,
                                              unsigned lds_addr) {
  uint64_t ga = (uint64_t)(uintptr_t)gptr;
  u32x4 g0;
  g0.x = 1u;                                    // count=1, is_restore=0, gather=0
  g0.y = lds_addr;                              // bits 63:32  : LDS byte address
  g0.z = (unsigned)ga;                          // bits 95:64  : global_addr[31:0]
  g0.w = (unsigned)(ga >> 32) | 0x80000000u;    // global_addr[56:32] | type=2
  i32x8 g1;
  g1[0] = (1 << 16) | (1 << 20) | (3 << 22) | (3 << 25); // data_size=2B, pad_en, 16dw->pad 4dw
  g1[1] = (dim0 & 0xFFFF) << 16;                // atomic_barrier_addr=0 | tensor_dim0 lo16
  g1[2] = (dim0 >> 16) | ((DH & 0xFFFF) << 16); // tensor_dim0 hi16 | tensor_dim1 lo16
  g1[3] = (32 << 16);                           // tensor_dim1 hi16=0 | tile_dim0=32
  g1[4] = DH;                                   // tile_dim1=128, tile_dim2=0 (2D)
  g1[5] = (int)stride;                          // tensor_dim0_stride lo32 (elements)
  g1[6] = 0;                                    // stride hi16 | tensor_dim1_stride lo16
  g1[7] = 0;
  i32x4 gz = {0, 0, 0, 0};
#if TDM_6ARG
  i32x8 gz8 = {0, 0, 0, 0, 0, 0, 0, 0};
  __builtin_amdgcn_tensor_load_to_lds(g0, g1, gz, gz, gz8, 0);
#else
  __builtin_amdgcn_tensor_load_to_lds(g0, g1, gz, gz, 0);
#endif
}

// ---------------- conversion kernels ----------------
__global__ void k_f32_to_bf16(const float* __restrict__ in, unsigned short* __restrict__ out, int n) {
  int i = blockIdx.x * blockDim.x + threadIdx.x;
  if (i < n) out[i] = f2bf(in[i]);
}

// W [F,D] f32 -> Wt [D,F] bf16 (row stride F so TDM tiles are [D rows x K cols])
__global__ void k_wt_bf16(const float* __restrict__ W, unsigned short* __restrict__ Wt, int F, int D) {
  int i = blockIdx.x * blockDim.x + threadIdx.x;
  if (i < F * D) { int f = i / D, d = i % D; Wt[(size_t)d * F + f] = f2bf(W[i]); }
}

// ---------------- h = x@W + b  (bf16 WMMA, f32 accum) ----------------
// TDM double-buffers the 32x128 Wt tile in LDS (shared by all 8 waves); the
// next K-step's tile streams in behind this step's WMMAs (s_wait_tensorcnt +
// barrier instead of per-tile s_wait_loadcnt 0). A operand streams from global.
__global__ void k_gemm1(const unsigned short* __restrict__ Xb,  // [ROWS,F] bf16
                        const unsigned short* __restrict__ Wt,  // [DH,F]  bf16
                        const float* __restrict__ bias,         // [DH]
                        float* __restrict__ H,                  // [ROWS,DH] f32
                        unsigned short* __restrict__ Ht,        // [NB,DH,NN] bf16
                        int F) {
  __shared__ __align__(16) char smem[2 * BUF_BYTES];
  const int lane = threadIdx.x & 31;
  const int wave = threadIdx.x >> 5;
  const int g = lane >> 4;
  const int r = lane & 15;
  const int rowbase = blockIdx.x * 128 + wave * 16;
  const unsigned lds0 = (unsigned)(uintptr_t)smem;

  if (wave == 0) tdm_tile_load(Wt + 0, F, F, lds0);       // prefetch kb=0 -> buf0

  v8f acc[8] = {};
  for (int kb = 0; kb < F; kb += 32) {
    const int cur = (kb >> 5) & 1;
    if (wave == 0) __builtin_amdgcn_s_wait_tensorcnt(0);  // cur tile landed
    __syncthreads();
    if (wave == 0 && kb + 32 < F)
      tdm_tile_load(Wt + kb + 32, F, F, lds0 + (cur ^ 1) * BUF_BYTES);

    // B tiles from LDS first (ds latency hides behind the A global load wait)
    const unsigned short* bufc = (const unsigned short*)(smem + cur * BUF_BYTES);
    ABu Bv[8];
#pragma unroll
    for (int t = 0; t < 8; t++) {
      const unsigned short* hp = bufc + (t * 16 + r) * TILE_PITCH_H + g * 8;
      Bv[t].q[0] = *(const uint4*)(hp);
      Bv[t].q[1] = *(const uint4*)(hp + 16);
    }
    ABu A;
    const unsigned short* xr = Xb + (size_t)(rowbase + r) * F + kb + g * 8;
    A.q[0] = *(const uint4*)(xr);
    A.q[1] = *(const uint4*)(xr + 16);
#pragma unroll
    for (int t = 0; t < 8; t++)
      acc[t] = __builtin_amdgcn_wmma_f32_16x16x32_bf16(false, A.v, false, Bv[t].v,
                                                       (short)0, acc[t], false, false);
  }
#pragma unroll
  for (int t = 0; t < 8; t++) {
    float bv = bias[t * 16 + r];
#pragma unroll
    for (int v = 0; v < 8; v++) {
      int row = rowbase + v + 8 * g;   // C: M = v + 8g, N = lane&15
      int d = t * 16 + r;
      float val = acc[t][v] + bv;
      H[(size_t)row * DH + d] = val;
      int bb = row >> 11, nn = row & (NN - 1);
      Ht[((size_t)bb * DH + d) * NN + nn] = f2bf(val);
    }
  }
}

// ---------------- sl = h@a_l, sr = h@a_r ----------------
__global__ void k_score(const float* __restrict__ H, const float* __restrict__ a,
                        float* __restrict__ sl, float* __restrict__ sr) {
  int lane = threadIdx.x & 31, wave = threadIdx.x >> 5;
  int row = blockIdx.x * 8 + wave;
  const float* hr = H + (size_t)row * DH;
  float al = 0.f, ar = 0.f;
#pragma unroll
  for (int t = 0; t < 4; t++) {
    int d = lane + t * 32;
    float v = hr[d];
    al += v * a[d];
    ar += v * a[DH + d];
  }
  for (int off = 16; off; off >>= 1) {
    al += __shfl_down(al, off, 32);
    ar += __shfl_down(ar, off, 32);
  }
  if (lane == 0) { sl[row] = al; sr[row] = ar; }
}

// ---------------- per-batch top-2 of sr (masked row max via monotone lrelu) ----------------
__global__ void k_top2(const float* __restrict__ sr, float* __restrict__ m12, int* __restrict__ i1o) {
  __shared__ float sm1[256], sm2[256];
  __shared__ int si[256];
  int b = blockIdx.x, tid = threadIdx.x;
  float m1 = -1e30f, m2 = -1e30f; int i1 = -1;
  for (int n = tid; n < NN; n += 256) {
    float v = sr[(size_t)b * NN + n];
    if (v > m1) { m2 = m1; m1 = v; i1 = n; }
    else if (v > m2) m2 = v;
  }
  sm1[tid] = m1; sm2[tid] = m2; si[tid] = i1;
  __syncthreads();
  if (tid == 0) {
    for (int k = 1; k < 256; k++) {
      if (sm1[k] > m1) { m2 = fmaxf(m1, sm2[k]); m1 = sm1[k]; i1 = si[k]; }
      else { m2 = fmaxf(m2, sm1[k]); }
    }
    m12[b * 2] = m1; m12[b * 2 + 1] = m2; i1o[b] = i1;
  }
}

// ---------------- per-row softmax stats: max + 1/denominator ----------------
__global__ void k_rowstats(const float* __restrict__ sl, const float* __restrict__ sr,
                           const float* __restrict__ m12, const int* __restrict__ i1a,
                           float* __restrict__ rm, float* __restrict__ invd) {
  __shared__ float srs[NN];
  int gi = blockIdx.x * 256 + threadIdx.x;
  int b = gi >> 11;
  int i_loc = gi & (NN - 1);
  for (int k = threadIdx.x; k < NN; k += 256) srs[k] = sr[(size_t)b * NN + k];
  __syncthreads();
  float s_i = sl[gi];
  float m1 = m12[b * 2], m2 = m12[b * 2 + 1];
  int i1 = i1a[b];
  float mx = lrelu(s_i + ((i_loc == i1) ? m2 : m1));
  float acc = 0.f;
  for (int j = 0; j < NN; j++) acc += __expf(lrelu(s_i + srs[j]) - mx);
  acc -= __expf(lrelu(s_i + srs[i_loc]) - mx);   // masked diagonal
  rm[gi] = mx;
  invd[gi] = 1.f / acc;
}

// ---------------- out = elu(softmax(s)@h + h_i) ----------------
// TDM double-buffers the shared 32x128 Ht tile in LDS; B ds-loads issue first,
// the softmax-weight VALU block hides their latency, then 8 WMMAs run back-to-back.
__global__ void k_attn(const unsigned short* __restrict__ Ht,  // [NB,DH,NN] bf16
                       const float* __restrict__ H,            // [ROWS,DH] f32
                       const float* __restrict__ sl, const float* __restrict__ sr,
                       const float* __restrict__ rm, const float* __restrict__ invd,
                       float* __restrict__ out, int featOff) {
  __shared__ __align__(16) char smem[SR_BYTES + 2 * BUF_BYTES];
  float* srs = (float*)smem;
  const int lane = threadIdx.x & 31, wave = threadIdx.x >> 5;
  const int g = lane >> 4, r = lane & 15;
  const int rowbase = blockIdx.x * 128 + wave * 16;
  const int bb = rowbase >> 11;
  const int nloc0 = rowbase & (NN - 1);
  for (int k = threadIdx.x; k < NN; k += 256) srs[k] = sr[(size_t)bb * NN + k];

  const unsigned short* htb = Ht + (size_t)bb * DH * NN;  // this batch's [DH,NN]
  const unsigned lds0 = (unsigned)(uintptr_t)(smem + SR_BYTES);

  if (wave == 0) tdm_tile_load(htb + 0, NN, NN, lds0);    // prefetch jb=0 -> buf0

  const int rowg = rowbase + r;
  const int row_loc = nloc0 + r;
  const float s_i = sl[rowg];
  const float mx = rm[rowg];
  const float iv = invd[rowg];

  v8f acc[8] = {};
  for (int jb = 0; jb < NN; jb += 32) {
    const int cur = (jb >> 5) & 1;
    if (wave == 0) __builtin_amdgcn_s_wait_tensorcnt(0);  // cur buffer landed
    __syncthreads();                                      // broadcast readiness
    if (wave == 0 && jb + 32 < NN)
      tdm_tile_load(htb + jb + 32, NN, NN, lds0 + (cur ^ 1) * BUF_BYTES);

    // issue all B ds-loads first ...
    const unsigned short* bufc =
        (const unsigned short*)(smem + SR_BYTES + cur * BUF_BYTES);
    ABu Bv[8];
#pragma unroll
    for (int t = 0; t < 8; t++) {
      const unsigned short* hp = bufc + (t * 16 + r) * TILE_PITCH_H + g * 8;
      Bv[t].q[0] = *(const uint4*)(hp);       // ds_load_b128
      Bv[t].q[1] = *(const uint4*)(hp + 16);
    }
    // ... softmax-weight VALU block hides DS latency ...
    ABu A;
#pragma unroll
    for (int e = 0; e < 16; e++) {
      int j = jb + ((e < 8) ? 0 : 16) + g * 8 + (e & 7);  // ISA A-layout K index
      float p = __expf(lrelu(s_i + srs[j]) - mx) * iv;
      if (j == row_loc) p = 0.f;                          // masked diagonal
      A.s[e] = f2bf(p);
    }
    // ... then the XDL burst
#pragma unroll
    for (int t = 0; t < 8; t++)
      acc[t] = __builtin_amdgcn_wmma_f32_16x16x32_bf16(false, A.v, false, Bv[t].v,
                                                       (short)0, acc[t], false, false);
  }
#pragma unroll
  for (int t = 0; t < 8; t++) {
#pragma unroll
    for (int v = 0; v < 8; v++) {
      int row = rowbase + v + 8 * g;
      int d = t * 16 + r;
      float val = acc[t][v] + H[(size_t)row * DH + d];    // + eye contribution
      val = (val > 0.f) ? val : (__expf(val) - 1.f);      // ELU
      out[(size_t)row * FD + featOff + d] = val;
    }
  }
}

// ---------------- host orchestration ----------------
extern "C" void kernel_launch(void* const* d_in, const int* in_sizes, int n_in,
                              void* d_out, int out_size, void* d_ws, size_t ws_size,
                              hipStream_t stream) {
  const float* x = (const float*)d_in[0];
  // d_in[1] = edges (unused by the reference)
  const float* Wp[2][2] = {{(const float*)d_in[2], (const float*)d_in[5]},
                           {(const float*)d_in[8], (const float*)d_in[11]}};
  const float* bp[2][2] = {{(const float*)d_in[3], (const float*)d_in[6]},
                           {(const float*)d_in[9], (const float*)d_in[12]}};
  const float* ap[2][2] = {{(const float*)d_in[4], (const float*)d_in[7]},
                           {(const float*)d_in[10], (const float*)d_in[13]}};

  char* w = (char*)d_ws;
  auto alloc = [&](size_t bytes) -> void* {
    void* p = (void*)w;
    w += (bytes + 255) & ~(size_t)255;
    return p;
  };
  unsigned short* Xb = (unsigned short*)alloc((size_t)ROWS * FD * 2);
  float* Y = (float*)alloc((size_t)ROWS * FD * 4);
  float* Hh[2]; unsigned short* Htb[2]; unsigned short* Wt[2];
  float *slb[2], *srb[2];
  for (int h = 0; h < 2; h++) {
    Hh[h]  = (float*)alloc((size_t)ROWS * DH * 4);
    Htb[h] = (unsigned short*)alloc((size_t)ROWS * DH * 2);
    Wt[h]  = (unsigned short*)alloc((size_t)DH * FD * 2);
    slb[h] = (float*)alloc((size_t)ROWS * 4);
    srb[h] = (float*)alloc((size_t)ROWS * 4);
  }
  float* m12 = (float*)alloc(NB * 2 * 4);
  int*   i1b = (int*)alloc(NB * 4);
  float* rmb = (float*)alloc((size_t)ROWS * 4);
  float* ivb = (float*)alloc((size_t)ROWS * 4);

  for (int layer = 0; layer < 2; layer++) {
    const float* Xf = layer ? (const float*)Y : x;
    float* Yo = layer ? (float*)d_out : Y;
    int nX = ROWS * FD;
    k_f32_to_bf16<<<(nX + 255) / 256, 256, 0, stream>>>(Xf, Xb, nX);
    for (int h = 0; h < 2; h++) {
      k_wt_bf16<<<(FD * DH + 255) / 256, 256, 0, stream>>>(Wp[layer][h], Wt[h], FD, DH);
      k_gemm1<<<ROWS / 128, 256, 0, stream>>>(Xb, Wt[h], bp[layer][h], Hh[h], Htb[h], FD);
      k_score<<<ROWS / 8, 256, 0, stream>>>(Hh[h], ap[layer][h], slb[h], srb[h]);
      k_top2<<<NB, 256, 0, stream>>>(srb[h], m12, i1b);
      k_rowstats<<<ROWS / 256, 256, 0, stream>>>(slb[h], srb[h], m12, i1b, rmb, ivb);
      k_attn<<<ROWS / 128, 256, 0, stream>>>(Htb[h], Hh[h], slb[h], srb[h], rmb, ivb, Yo, h * DH);
    }
  }
}
